// TreeLSTMCell_63917703299457
// MI455X (gfx1250) — matrix operand order repaired and checked by reading, not compiled, not run on hardware
//
#include <hip/hip_runtime.h>
#include <hip/hip_bf16.h>
#include <math.h>

#define HDIM 256
#define KCH 4

typedef __attribute__((ext_vector_type(16))) __bf16 v16bf;
typedef __attribute__((ext_vector_type(8)))  __bf16 bf16x8;
typedef __attribute__((ext_vector_type(8)))  float  v8f;

union Frag { v16bf v; bf16x8 h[2]; };

__device__ __forceinline__ v8f wmma_bf16(Frag a, Frag b, v8f c) {
  // D = A(16x32) x B(32x16) + C, f32 accumulate
  return __builtin_amdgcn_wmma_f32_16x16x32_bf16(false, a.v, false, b.v,
                                                 (short)0, c, false, false);
}

#define LOG2E 1.4426950408889634f

// fast hardware-transcendental gates: v_exp_f32 + v_rcp_f32
__device__ __forceinline__ float fast_sigmoid(float x) {
  return __builtin_amdgcn_rcpf(1.0f + __builtin_amdgcn_exp2f(-x * LOG2E));
}
__device__ __forceinline__ float fast_tanh(float x) {
  // tanh(x) = 1 - 2/(exp(2x)+1); saturates correctly for large |x|
  return 1.0f - 2.0f * __builtin_amdgcn_rcpf(
                           1.0f + __builtin_amdgcn_exp2f(x * (2.0f * LOG2E)));
}

__device__ __forceinline__ void st_bf4(__bf16* p, float4 v) {
  p[0] = (__bf16)v.x; p[1] = (__bf16)v.y; p[2] = (__bf16)v.z; p[3] = (__bf16)v.w;
}

__device__ __forceinline__ float4 add4(float4 a, float4 b) {
  return make_float4(a.x + b.x, a.y + b.y, a.z + b.z, a.w + b.w);
}

__device__ __forceinline__ void lda(Frag& f, const __bf16* base, int ab) {
  f.h[0] = *(const bf16x8*)(base + ab);
  f.h[1] = *(const bf16x8*)(base + ab + 8);
}

__device__ __forceinline__ void ldb(Frag& b, const __bf16* p) {
  b.h[0] = *(const bf16x8*)(p);
  b.h[1] = *(const bf16x8*)(p + 8);
}

// ---------------------------------------------------------------------------
// Kernel 1: convert the 8 weight matrices (256x256 f32 each, row-major,
// contraction along the fast dim) to bf16 in d_ws (1 MB total).
// m order: 0:Wi 1:Wo 2:Wu 3:Wf 4:Ui 5:Uo 6:Uu 7:Uf
// ---------------------------------------------------------------------------
__global__ __launch_bounds__(256) void wconv_kernel(
    const float* __restrict__ Wi, const float* __restrict__ Wo,
    const float* __restrict__ Wu, const float* __restrict__ Wf,
    const float* __restrict__ Ui, const float* __restrict__ Uo,
    const float* __restrict__ Uu, const float* __restrict__ Uf,
    __bf16* __restrict__ out)
{
  int idx = blockIdx.x * 256 + threadIdx.x;   // 131072 threads, 4 elems each
  int m = idx >> 14;                          // 16384 float4 per matrix
  int e = (idx & 16383) << 2;
  const float* src;
  switch (m) {
    case 0: src = Wi; break; case 1: src = Wo; break;
    case 2: src = Wu; break; case 3: src = Wf; break;
    case 4: src = Ui; break; case 5: src = Uo; break;
    case 6: src = Uu; break; default: src = Uf; break;
  }
  float4 f = *(const float4*)(src + e);
  st_bf4(out + (m << 16) + e, f);
}

// ---------------------------------------------------------------------------
// Kernel 2: block = 16 rows, 8 waves; A-side (x, h_sum, 4 children) is
// converted to bf16 in LDS exactly ONCE per row; each wave computes two
// 16-col tiles, streaming bf16 B-fragments from L2 in 16-wide load clauses.
// ---------------------------------------------------------------------------
__global__ __launch_bounds__(256) void treelstm_kernel(
    const float* __restrict__ x, const float* __restrict__ child_h,
    const float* __restrict__ child_c,
    const float* __restrict__ bi_g, const float* __restrict__ bf_g,
    const float* __restrict__ bo_g, const float* __restrict__ bu_g,
    const __bf16* __restrict__ wbf,
    float* __restrict__ out, int N)
{
  __shared__ __attribute__((aligned(16))) __bf16 lds_x [16 * HDIM];
  __shared__ __attribute__((aligned(16))) __bf16 lds_hs[16 * HDIM];
  __shared__ __attribute__((aligned(16))) __bf16 lds_ch[KCH][16 * HDIM];

  const int tid = threadIdx.x;
  const int r0  = blockIdx.x * 16;

  // ---- stage: x rows, child_h rows (bf16) and f32-accumulated h_sum ----
  {
    const int row = tid >> 4;              // 0..15
    const int cb  = (tid & 15) * 16;       // 16 contiguous columns
    const int n   = r0 + row;
    const int nc  = (n < N) ? n : (N - 1); // clamp; stores guarded later
    const float4* gx = (const float4*)(x + (size_t)nc * HDIM + cb);
    float4 x0 = gx[0], x1 = gx[1], x2 = gx[2], x3 = gx[3];
    __bf16* px = &lds_x[row * HDIM + cb];
    st_bf4(px + 0, x0); st_bf4(px + 4, x1); st_bf4(px + 8, x2); st_bf4(px + 12, x3);

    float4 h0, h1, h2, h3;
#pragma unroll
    for (int k = 0; k < KCH; ++k) {
      const float4* gc =
          (const float4*)(child_h + ((size_t)nc * KCH + k) * HDIM + cb);
      float4 c0 = gc[0], c1 = gc[1], c2 = gc[2], c3 = gc[3];
      if (k == 0) { h0 = c0; h1 = c1; h2 = c2; h3 = c3; }
      else { h0 = add4(h0, c0); h1 = add4(h1, c1); h2 = add4(h2, c2); h3 = add4(h3, c3); }
      __bf16* pc = &lds_ch[k][row * HDIM + cb];
      st_bf4(pc + 0, c0); st_bf4(pc + 4, c1); st_bf4(pc + 8, c2); st_bf4(pc + 12, c3);
    }
    __bf16* ph = &lds_hs[row * HDIM + cb];
    st_bf4(ph + 0, h0); st_bf4(ph + 4, h1); st_bf4(ph + 8, h2); st_bf4(ph + 12, h3);
  }
  __syncthreads();

  const int lane = tid & 31;
  const int lo   = lane & 15;
  const int hi   = lane >> 4;
  const int wave = tid >> 5;

#pragma unroll 1
  for (int t = 0; t < 2; ++t) {
    const int j0 = (wave * 2 + t) * 16;

    v8f z8 = {};
    v8f acc_i = z8, acc_o = z8, acc_u = z8, acc_fx = z8;
    v8f acc_f0 = z8, acc_f1 = z8, acc_f2 = z8, acc_f3 = z8;

#pragma unroll 1
    for (int k0 = 0; k0 < HDIM; k0 += 32) {
      // B fragments: one clause of 16 global_load_b128 from L2-resident ws
      const int bb = (j0 + lo) * HDIM + k0 + hi * 16;
      Frag b0, b1, b2, b3, b4, b5, b6, b7;
      ldb(b0, &wbf[(0 << 16) + bb]);
      ldb(b1, &wbf[(1 << 16) + bb]);
      ldb(b2, &wbf[(2 << 16) + bb]);
      ldb(b3, &wbf[(3 << 16) + bb]);
      ldb(b4, &wbf[(4 << 16) + bb]);
      ldb(b5, &wbf[(5 << 16) + bb]);
      ldb(b6, &wbf[(6 << 16) + bb]);
      ldb(b7, &wbf[(7 << 16) + bb]);

      // A fragments from LDS (staged once per row)
      const int ab = lo * HDIM + k0 + hi * 16;
      Frag ax, ah, a0, a1, a2, a3;
      lda(ax, lds_x, ab);
      lda(ah, lds_hs, ab);
      lda(a0, lds_ch[0], ab);
      lda(a1, lds_ch[1], ab);
      lda(a2, lds_ch[2], ab);
      lda(a3, lds_ch[3], ab);

      acc_i  = wmma_bf16(ax, b0, acc_i);   // x @ Wi^T
      acc_o  = wmma_bf16(ax, b1, acc_o);   // x @ Wo^T
      acc_u  = wmma_bf16(ax, b2, acc_u);   // x @ Wu^T
      acc_fx = wmma_bf16(ax, b3, acc_fx);  // x @ Wf^T
      acc_i  = wmma_bf16(ah, b4, acc_i);   // h_sum @ Ui^T
      acc_o  = wmma_bf16(ah, b5, acc_o);   // h_sum @ Uo^T
      acc_u  = wmma_bf16(ah, b6, acc_u);   // h_sum @ Uu^T
      acc_f0 = wmma_bf16(a0, b7, acc_f0);  // h_k @ Uf^T
      acc_f1 = wmma_bf16(a1, b7, acc_f1);
      acc_f2 = wmma_bf16(a2, b7, acc_f2);
      acc_f3 = wmma_bf16(a3, b7, acc_f3);
    }

    // ---- epilogue: biases, gates, child_c combine (f32, HW transcendentals) ----
    const int col = j0 + lo;
    const float bis = bi_g[col], bos = bo_g[col];
    const float bus = bu_g[col], bfs = bf_g[col];
    float* out_h = out;
    float* out_c = out + (size_t)N * HDIM;

#pragma unroll
    for (int v = 0; v < 8; ++v) {
      const int n = r0 + v + hi * 8;      // C/D layout: VGPR v -> row v (+8)
      if (n < N) {
        const float iv = fast_sigmoid(acc_i[v] + bis);
        const float ov = fast_sigmoid(acc_o[v] + bos);
        const float uv = fast_tanh(acc_u[v] + bus);
        const float pf = acc_fx[v] + bfs;
        const float* cc = child_c + (size_t)n * (KCH * HDIM) + col;
        float c = iv * uv;
        c += fast_sigmoid(pf + acc_f0[v]) * cc[0 * HDIM];
        c += fast_sigmoid(pf + acc_f1[v]) * cc[1 * HDIM];
        c += fast_sigmoid(pf + acc_f2[v]) * cc[2 * HDIM];
        c += fast_sigmoid(pf + acc_f3[v]) * cc[3 * HDIM];
        out_c[(size_t)n * HDIM + col] = c;
        out_h[(size_t)n * HDIM + col] = ov * fast_tanh(c);
      }
    }
  }
}

extern "C" void kernel_launch(void* const* d_in, const int* in_sizes, int n_in,
                              void* d_out, int out_size, void* d_ws, size_t ws_size,
                              hipStream_t stream) {
  const float* x       = (const float*)d_in[0];
  const float* child_h = (const float*)d_in[1];
  const float* child_c = (const float*)d_in[2];
  const float* Wi = (const float*)d_in[3];
  const float* bi = (const float*)d_in[4];
  const float* Wf = (const float*)d_in[5];
  const float* bf = (const float*)d_in[6];
  const float* Wo = (const float*)d_in[7];
  const float* bo = (const float*)d_in[8];
  const float* Wu = (const float*)d_in[9];
  const float* bu = (const float*)d_in[10];
  const float* Ui = (const float*)d_in[11];
  const float* Uf = (const float*)d_in[12];
  const float* Uo = (const float*)d_in[13];
  const float* Uu = (const float*)d_in[14];

  const int N = in_sizes[0] / HDIM;          // 100000
  __bf16* wbf = (__bf16*)d_ws;               // 8 * 65536 bf16 = 1 MB scratch

  wconv_kernel<<<512, 256, 0, stream>>>(Wi, Wo, Wu, Wf, Ui, Uo, Uu, Uf, wbf);
  treelstm_kernel<<<(N + 15) / 16, 256, 0, stream>>>(
      x, child_h, child_c, bi, bf, bo, bu, wbf, (float*)d_out, N);
}
